// DeepTypedGraphNet_71665824301916
// MI455X (gfx1250) — matrix operand order repaired and checked by reading, not compiled, not run on hardware
//
#include <hip/hip_runtime.h>

#define NN 50000
#define NE 800000

typedef __attribute__((ext_vector_type(16))) __bf16 v16bf;
typedef __attribute__((ext_vector_type(8)))  float  v8f;

// ---------------------------------------------------------------------------
// A-fragment loaders (ISA 16x32 bf16 A layout):
//   lane<16 : row = lane,    halves 0-7 -> K=k0..k0+7,   8-15 -> K=k0+16..+23
//   lane>=16: row = lane-16, halves 0-7 -> K=k0+8..+15,  8-15 -> K=k0+24..+31
// caller passes p = rowptr + ktile*32 + hi8, so runs are p[0..7] and p[16..23]
// ---------------------------------------------------------------------------
static __device__ __forceinline__ v16bf load_a_fast(const float* p) {
  v16bf a;
#pragma unroll
  for (int i = 0; i < 8; ++i) {
    a[i]     = (__bf16)p[i];
    a[i + 8] = (__bf16)p[16 + i];
  }
  return a;
}

static __device__ __forceinline__ v16bf load_a_bounded(const float* row, int kbase,
                                                       int hi8, int lim) {
  v16bf a;
#pragma unroll
  for (int i = 0; i < 8; ++i) {
    int k0 = kbase + hi8 + i;
    int k1 = kbase + hi8 + 16 + i;
    a[i]     = (__bf16)((k0 < lim) ? row[k0] : 0.0f);
    a[i + 8] = (__bf16)((k1 < lim) ? row[k1] : 0.0f);
  }
  return a;
}

// B fragments pre-packed: offset ((kt*NT + nt)*32 + lane)*16 bf16 (32B/lane).
template <int NT>
static __device__ __forceinline__ void wmma_k(v8f* acc, v16bf a, const __bf16* wf,
                                              int kt, int lane) {
#pragma unroll
  for (int nt = 0; nt < NT; ++nt) {
    v16bf b = *(const v16bf*)(wf + (((size_t)(kt * NT + nt)) * 32 + lane) * 16);
    acc[nt] = __builtin_amdgcn_wmma_f32_16x16x32_bf16(false, a, false, b, (short)0,
                                                      acc[nt], false, false);
  }
}

// layer-1 epilogue: bias + SiLU, C-layout regs -> LDS tile [16][72]
static __device__ __forceinline__ void silu_to_lds(float* hbuf, const v8f* acc,
                                                   const float* b0, int lane) {
  int hi8 = (lane >= 16) ? 8 : 0, colb = lane & 15;
#pragma unroll
  for (int nt = 0; nt < 4; ++nt) {
    int col = nt * 16 + colb;
    float bb = b0[col];
#pragma unroll
    for (int r = 0; r < 8; ++r) {
      float x = acc[nt][r] + bb;
      hbuf[(r + hi8) * 72 + col] = x / (1.0f + __expf(-x));
    }
  }
}

// layer-2: preload both LDS A-fragments, then 8 WMMAs
static __device__ __forceinline__ void gemm2(v8f* acc, const float* hrow, int hi8,
                                             const __bf16* w1f, int lane) {
  v16bf a0 = load_a_fast(hrow + 0 * 32 + hi8);
  v16bf a1 = load_a_fast(hrow + 1 * 32 + hi8);
  wmma_k<4>(acc, a0, w1f, 0, lane);
  wmma_k<4>(acc, a1, w1f, 1, lane);
}

// layer-2 epilogue: bias -> LDS, rowwise LayerNorm, optional residual,
// write f32 out; optional f32 atomic scatter-add into agg[receivers[row]].
static __device__ __forceinline__ void ln_out(float* ybuf, const v8f* acc,
                                              const float* b1, const float* g,
                                              const float* beta, const float* resid,
                                              float* out, int mtile, int lane,
                                              float* agg, const int* receivers) {
  int hi8 = (lane >= 16) ? 8 : 0, colb = lane & 15;
#pragma unroll
  for (int nt = 0; nt < 4; ++nt) {
    int col = nt * 16 + colb;
    float bb = b1[col];
#pragma unroll
    for (int r = 0; r < 8; ++r) ybuf[(r + hi8) * 72 + col] = acc[nt][r] + bb;
  }
  int lrow = lane & 15;
  const float* yr = ybuf + lrow * 72;
  int c0 = (lane >= 16) ? 32 : 0;
  float s = 0.f, sq = 0.f;
#pragma unroll
  for (int c = 0; c < 32; ++c) {
    float v = yr[c0 + c];
    s += v;
    sq += v * v;
  }
  s += __shfl_xor(s, 16, 32);
  sq += __shfl_xor(sq, 16, 32);
  float mean = s * (1.f / 64.f);
  float var = sq * (1.f / 64.f) - mean * mean;
  float rs = rsqrtf(var + 1e-5f);
  int orow = mtile * 16 + lrow;
  int ridx = (receivers != nullptr) ? receivers[orow] : 0;
#pragma unroll
  for (int c = 0; c < 32; ++c) {
    int col = c0 + c;
    float v = (yr[col] - mean) * rs * g[col] + beta[col];
    if (resid) v += resid[(size_t)orow * 64 + col];
    out[(size_t)orow * 64 + col] = v;
    if (agg) atomicAdd(&agg[(size_t)ridx * 64 + col], v);
  }
}

// ---------------------------------------------------------------------------
// Fused MLP+LN over up to two f32 input blocks. Compile-time k-tile counts so
// all GEMM loops fully unroll; all A fragments preloaded into registers before
// the WMMA chain so B-fragment loads overlap matrix ops.
// One wave per 16-row tile, 4 waves / block.
// ---------------------------------------------------------------------------
template <int KT1, int KT2, bool BOUND1>
__global__ void __launch_bounds__(128) fused_mlp(
    const float* X1, int ld1, int Din1, const float* X2, const __bf16* w0f,
    const float* b0, const __bf16* w1f, const float* b1, const float* g,
    const float* beta, const float* resid, float* out, int mtiles) {
  __shared__ float smem[4][16 * 72];
  int wave = threadIdx.x >> 5, lane = threadIdx.x & 31;
  int mtile = blockIdx.x * 4 + wave;
  if (mtile >= mtiles) return;
  float* hbuf = smem[wave];
  int lrow = lane & 15, hi8 = (lane >= 16) ? 8 : 0;
  const float* xrow = X1 + (size_t)(mtile * 16 + lrow) * ld1;

  v16bf afr[KT1 + KT2];
#pragma unroll
  for (int kt = 0; kt < KT1; ++kt) {
    if (BOUND1)
      afr[kt] = load_a_bounded(xrow, kt * 32, hi8, Din1);
    else
      afr[kt] = load_a_fast(xrow + kt * 32 + hi8);
  }
  if constexpr (KT2 > 0) {
    const float* x2row = X2 + (size_t)(mtile * 16 + lrow) * 64;
#pragma unroll
    for (int kt = 0; kt < KT2; ++kt)
      afr[KT1 + kt] = load_a_fast(x2row + kt * 32 + hi8);
  }

  v8f acc0[4] = {};
#pragma unroll
  for (int kt = 0; kt < KT1 + KT2; ++kt) wmma_k<4>(acc0, afr[kt], w0f, kt, lane);

  silu_to_lds(hbuf, acc0, b0, lane);

  v8f acc1[4] = {};
  gemm2(acc1, hbuf + lrow * 72, hi8, w1f, lane);

  ln_out(hbuf, acc1, b1, g, beta, resid, out, mtile, lane, nullptr, nullptr);
}

// Edge block: A = concat(e[row], n[senders[row]], n[receivers[row]]) (192),
// MLP+LN, residual into e (in place), f32 atomic scatter-add into agg.
__global__ void __launch_bounds__(128) edge_block_kernel(
    float* e, const float* n, const int* senders, const int* receivers,
    const __bf16* w0f, const float* b0, const __bf16* w1f, const float* b1,
    const float* g, const float* beta, float* agg, int mtiles) {
  __shared__ float smem[4][16 * 72];
  int wave = threadIdx.x >> 5, lane = threadIdx.x & 31;
  int mtile = blockIdx.x * 4 + wave;
  if (mtile >= mtiles) return;
  float* hbuf = smem[wave];
  int lrow = lane & 15, hi8 = (lane >= 16) ? 8 : 0;
  int row = mtile * 16 + lrow;
  const float* erow = e + (size_t)row * 64;
  const float* srow = n + (size_t)senders[row] * 64;
  const float* rrow = n + (size_t)receivers[row] * 64;

  v16bf afr[6];
#pragma unroll
  for (int kt = 0; kt < 2; ++kt) {
    afr[kt]     = load_a_fast(erow + kt * 32 + hi8);
    afr[2 + kt] = load_a_fast(srow + kt * 32 + hi8);
    afr[4 + kt] = load_a_fast(rrow + kt * 32 + hi8);
  }

  v8f acc0[4] = {};
#pragma unroll
  for (int kt = 0; kt < 6; ++kt) wmma_k<4>(acc0, afr[kt], w0f, kt, lane);

  silu_to_lds(hbuf, acc0, b0, lane);

  v8f acc1[4] = {};
  gemm2(acc1, hbuf + lrow * 72, hi8, w1f, lane);

  ln_out(hbuf, acc1, b1, g, beta, e, e, mtile, lane, agg, receivers);
}

// Decoder: n[64] -> SiLU MLP -> 16, no LN.
__global__ void __launch_bounds__(128) decoder_kernel(
    const float* n, const __bf16* w0f, const float* b0, const __bf16* w1f,
    const float* b1, float* out, int mtiles) {
  __shared__ float smem[4][16 * 72];
  int wave = threadIdx.x >> 5, lane = threadIdx.x & 31;
  int mtile = blockIdx.x * 4 + wave;
  if (mtile >= mtiles) return;
  float* hbuf = smem[wave];
  int lrow = lane & 15, hi8 = (lane >= 16) ? 8 : 0;
  const float* nrow = n + (size_t)(mtile * 16 + lrow) * 64;

  v16bf a0 = load_a_fast(nrow + 0 * 32 + hi8);
  v16bf a1 = load_a_fast(nrow + 1 * 32 + hi8);
  v8f acc0[4] = {};
  wmma_k<4>(acc0, a0, w0f, 0, lane);
  wmma_k<4>(acc0, a1, w0f, 1, lane);
  silu_to_lds(hbuf, acc0, b0, lane);

  const float* hrow = hbuf + lrow * 72;
  v16bf h0 = load_a_fast(hrow + 0 * 32 + hi8);
  v16bf h1 = load_a_fast(hrow + 1 * 32 + hi8);
  v8f acc1[1] = {};
  wmma_k<1>(acc1, h0, w1f, 0, lane);
  wmma_k<1>(acc1, h1, w1f, 1, lane);

  int colb = lane & 15;
  float bb = b1[colb];
#pragma unroll
  for (int r = 0; r < 8; ++r)
    out[(size_t)(mtile * 16 + r + hi8) * 16 + colb] = acc1[0][r] + bb;
}

// Pack f32 weights [Kdim, Ncols] into bf16 WMMA B-fragment layout
// (zero-padded to KT*32 rows): dst[((kt*NT+nt)*32+lane)*16 + h],
// K = kt*32 + (lane>=16 ? 8:0) + (h&7) + (h>=8 ? 16:0), col = nt*16 + (lane&15).
__global__ void pack_w(const float* w, __bf16* dst, int Kdim, int KT, int NT,
                       int Ncols) {
  int idx = blockIdx.x * 256 + threadIdx.x;
  int total = KT * NT * 512;
  if (idx >= total) return;
  int h = idx & 15;
  int lane = (idx >> 4) & 31;
  int ktnt = idx >> 9;
  int nt = ktnt % NT;
  int kt = ktnt / NT;
  int k = kt * 32 + ((lane >= 16) ? 8 : 0) + (h & 7) + ((h >= 8) ? 16 : 0);
  int col = nt * 16 + (lane & 15);
  float v = (k < Kdim) ? w[(size_t)k * Ncols + col] : 0.0f;
  dst[idx] = (__bf16)v;
}

__global__ void zero_f32(float* p, int n) {
  int i = blockIdx.x * 256 + threadIdx.x;
  if (i < n) p[i] = 0.0f;
}

extern "C" void kernel_launch(void* const* d_in, const int* in_sizes, int n_in,
                              void* d_out, int out_size, void* d_ws, size_t ws_size,
                              hipStream_t stream) {
  (void)in_sizes; (void)n_in; (void)out_size; (void)ws_size;
  const float* node_feat = (const float*)d_in[0];
  const float* edge_feat = (const float*)d_in[1];
  const int* senders = (const int*)d_in[2];
  const int* receivers = (const int*)d_in[3];
  const float* ne_w0 = (const float*)d_in[4];
  const float* ne_b0 = (const float*)d_in[5];
  const float* ne_w1 = (const float*)d_in[6];
  const float* ne_b1 = (const float*)d_in[7];
  const float* ne_g = (const float*)d_in[8];
  const float* ne_be = (const float*)d_in[9];
  const float* ee_w0 = (const float*)d_in[10];
  const float* ee_b0 = (const float*)d_in[11];
  const float* ee_w1 = (const float*)d_in[12];
  const float* ee_b1 = (const float*)d_in[13];
  const float* ee_g = (const float*)d_in[14];
  const float* ee_be = (const float*)d_in[15];
  const float* eb_w0 = (const float*)d_in[16];
  const float* eb_b0 = (const float*)d_in[17];
  const float* eb_w1 = (const float*)d_in[18];
  const float* eb_b1 = (const float*)d_in[19];
  const float* eb_g = (const float*)d_in[20];
  const float* eb_be = (const float*)d_in[21];
  const float* nb_w0 = (const float*)d_in[22];
  const float* nb_b0 = (const float*)d_in[23];
  const float* nb_w1 = (const float*)d_in[24];
  const float* nb_b1 = (const float*)d_in[25];
  const float* nb_g = (const float*)d_in[26];
  const float* nb_be = (const float*)d_in[27];
  const float* de_w0 = (const float*)d_in[28];
  const float* de_b0 = (const float*)d_in[29];
  const float* de_w1 = (const float*)d_in[30];
  const float* de_b1 = (const float*)d_in[31];

  char* base = (char*)d_ws;
  size_t off = 0;
  auto alloc = [&](size_t bytes) -> void* {
    void* p = base + off;
    off += (bytes + 255) & ~(size_t)255;
    return p;
  };
  float* nbuf = (float*)alloc((size_t)NN * 64 * 4);   // node latents A
  float* nbuf2 = (float*)alloc((size_t)NN * 64 * 4);  // node latents B
  float* ebuf = (float*)alloc((size_t)NE * 64 * 4);   // edge latents (in place)
  float* agg = (float*)alloc((size_t)NN * 64 * 4);    // segment-sum scratch

  auto pack = [&](const float* w, int Kdim, int Ncols) -> const __bf16* {
    int KT = (Kdim + 31) / 32, NT = Ncols / 16;
    __bf16* dst = (__bf16*)alloc((size_t)KT * NT * 512 * 2);
    int total = KT * NT * 512;
    pack_w<<<(total + 255) / 256, 256, 0, stream>>>(w, dst, Kdim, KT, NT, Ncols);
    return dst;
  };

  const __bf16* P_ne0 = pack(ne_w0, 16, 64);
  const __bf16* P_ne1 = pack(ne_w1, 64, 64);
  const __bf16* P_ee0 = pack(ee_w0, 8, 64);
  const __bf16* P_ee1 = pack(ee_w1, 64, 64);
  const __bf16* P_eb0[2], *P_eb1[2], *P_nb0[2], *P_nb1[2];
  for (int s = 0; s < 2; ++s) {
    P_eb0[s] = pack(eb_w0 + (size_t)s * 192 * 64, 192, 64);
    P_eb1[s] = pack(eb_w1 + (size_t)s * 64 * 64, 64, 64);
    P_nb0[s] = pack(nb_w0 + (size_t)s * 128 * 64, 128, 64);
    P_nb1[s] = pack(nb_w1 + (size_t)s * 64 * 64, 64, 64);
  }
  const __bf16* P_d0 = pack(de_w0, 64, 64);
  const __bf16* P_d1 = pack(de_w1, 64, 16);

  int ntiles = NN / 16;  // 3125
  int etiles = NE / 16;  // 50000
  int nblocks = (ntiles + 3) / 4;
  int eblocks = (etiles + 3) / 4;

  // encoders (single bounded k-tile: Din = 16 / 8, zero-padded to 32)
  fused_mlp<1, 0, true><<<nblocks, 128, 0, stream>>>(
      node_feat, 16, 16, nullptr, P_ne0, ne_b0, P_ne1, ne_b1, ne_g, ne_be,
      nullptr, nbuf, ntiles);
  fused_mlp<1, 0, true><<<eblocks, 128, 0, stream>>>(
      edge_feat, 8, 8, nullptr, P_ee0, ee_b0, P_ee1, ee_b1, ee_g, ee_be, nullptr,
      ebuf, etiles);

  // message-passing steps (node latents double-buffered)
  float* ncur = nbuf;
  float* nnext = nbuf2;
  for (int s = 0; s < 2; ++s) {
    zero_f32<<<(NN * 64 + 255) / 256, 256, 0, stream>>>(agg, NN * 64);
    edge_block_kernel<<<eblocks, 128, 0, stream>>>(
        ebuf, ncur, senders, receivers, P_eb0[s], eb_b0 + s * 64, P_eb1[s],
        eb_b1 + s * 64, eb_g + s * 64, eb_be + s * 64, agg, etiles);
    // node block: X = concat(n, agg), residual n -> nnext
    fused_mlp<2, 2, false><<<nblocks, 128, 0, stream>>>(
        ncur, 64, 64, agg, P_nb0[s], nb_b0 + s * 64, P_nb1[s], nb_b1 + s * 64,
        nb_g + s * 64, nb_be + s * 64, ncur, nnext, ntiles);
    float* t = ncur; ncur = nnext; nnext = t;
  }

  decoder_kernel<<<nblocks, 128, 0, stream>>>(ncur, P_d0, de_b0, P_d1, de_b1,
                                              (float*)d_out, ntiles);
}
